// quantLlamaAttention_80058190397937
// MI455X (gfx1250) — matrix operand order repaired and checked by reading, not compiled, not run on hardware
//
#include <hip/hip_runtime.h>

// ---------------------------------------------------------------------------
// quantLlamaAttention for MI455X (gfx1250, wave32, WMMA + async-to-LDS DMA)
// cast fp32 -> bf16 once; all 5 matmuls via v_wmma_f32_16x16x32_bf16 (fp32
// accum); flash-style attention (no SxS tensor); GEMM K-slabs double-buffered
// in LDS and staged with global_load_async_to_lds_b128 (ASYNCcnt).
// ---------------------------------------------------------------------------

#define HID 4096
#define SEQ 2048
#define NH 32
#define NKV 8
#define HD 128

typedef __attribute__((ext_vector_type(16))) __bf16 v16bf;
typedef __attribute__((ext_vector_type(8)))  float  v8f;

union FragAB { v16bf v; unsigned u[8]; };
union FragC  { v8f   v; float    f[8]; };

__device__ __forceinline__ unsigned short f2bf(float x) {
  unsigned u = __float_as_uint(x);
  unsigned r = u + 0x7FFFu + ((u >> 16) & 1u);   // round-to-nearest-even
  return (unsigned short)(r >> 16);
}
__device__ __forceinline__ float bf2f(unsigned short b) {
  return __uint_as_float(((unsigned)b) << 16);
}
// A frag k order used here (contiguous; B matches). VGPR j of lane (half = lane>>4) holds the
// bf16 pair starting at this K index.
__device__ __forceinline__ int a_pair_k(int j, int half) {
  return 16 * half + 2 * j;
}
// Low 32 bits of a generic pointer to LDS == byte offset within the
// workgroup's LDS allocation (flat-aperture encoding, ISA 10.2).
__device__ __forceinline__ unsigned lds_off(const void* p) {
  return (unsigned)(unsigned long long)p;
}
// GVS-form async DMA: LDS[vdst_lds + ...] = MEM[saddr + voff], 16B per lane,
// tracked by ASYNCcnt (ISA 10. / 15.18.3).
__device__ __forceinline__ void async_load_b128(unsigned lds, unsigned voff,
                                                const void* base) {
  asm volatile("global_load_async_to_lds_b128 %0, %1, %2"
               :: "v"(lds), "v"(voff), "s"(base) : "memory");
}

// ---------------------------------------------------------------------------
__global__ void cast_f32_to_bf16(const float* __restrict__ src,
                                 unsigned short* __restrict__ dst, int n4) {
  int i = blockIdx.x * blockDim.x + threadIdx.x;
  if (i >= n4) return;
  float4 f = ((const float4*)src)[i];
  ushort4 o;
  o.x = f2bf(f.x); o.y = f2bf(f.y); o.z = f2bf(f.z); o.w = f2bf(f.w);
  ((ushort4*)dst)[i] = o;
}

// ---------------------------------------------------------------------------
// Issue one 128x32 bf16 slab of A and of W into LDS via async DMA.
// chunk i (of 512 per tile) = 16B at row (i>>2), col (i&3)*8; each of the 256
// threads owns chunks {tid, tid+256} -> 4 async instructions per wave.
__device__ __forceinline__ void gemm_issue_slab(
    const unsigned short* __restrict__ A, const unsigned short* __restrict__ W,
    int K, int k0, int m0, int n0, unsigned asBase, unsigned bsBase, int tid) {
  const int r = tid >> 2, c = tid & 3;
  const unsigned offA = (unsigned)((((m0 + r) * K) + k0 + c * 8) * 2);
  const unsigned offB = (unsigned)((((n0 + r) * K) + k0 + c * 8) * 2);
  const unsigned rows64 = (unsigned)(64 * K * 2);
  async_load_b128(asBase + tid * 16,           offA,          A);
  async_load_b128(asBase + (tid + 256) * 16,   offA + rows64, A);
  async_load_b128(bsBase + tid * 16,           offB,          W);
  async_load_b128(bsBase + (tid + 256) * 16,   offB + rows64, W);
}

// C[m,n] = sum_k A[m,k] * W[n,k]   (out = A @ W^T)
// A: [M,K] bf16 row-major, W: [N,K] bf16 row-major.
// 128x128 block tile, 8 waves (2x4), each wave 4x2 WMMA tiles.
// K staged in 32-wide slabs, double-buffered, async-DMA'd into LDS.
template <bool OUT_BF16>
__global__ __launch_bounds__(256)
void gemm_a_x_wT(const unsigned short* __restrict__ A,
                 const unsigned short* __restrict__ W,
                 unsigned short* __restrict__ Obf,
                 float* __restrict__ Ofp,
                 int K, int ldo) {
  const int tid  = threadIdx.x;
  const int wave = tid >> 5;
  const int lane = tid & 31;
  const int half = lane >> 4;
  const int l15  = lane & 15;
  const int wr   = wave >> 2;           // 0..1 -> 64 M-rows
  const int wc   = wave & 3;            // 0..3 -> 32 N-cols
  const int m0   = blockIdx.y * 128;
  const int n0   = blockIdx.x * 128;

  __shared__ unsigned short As[2][128][32];   // [buf][m][k]
  __shared__ unsigned short Bs[2][128][32];   // [buf][n][k]
  const unsigned asB[2] = { lds_off(&As[0][0][0]), lds_off(&As[1][0][0]) };
  const unsigned bsB[2] = { lds_off(&Bs[0][0][0]), lds_off(&Bs[1][0][0]) };

  FragC acc[4][2];
#pragma unroll
  for (int mt = 0; mt < 4; ++mt)
#pragma unroll
    for (int nt = 0; nt < 2; ++nt)
#pragma unroll
      for (int r = 0; r < 8; ++r) acc[mt][nt].f[r] = 0.f;

  const int nslab = K / 32;
  gemm_issue_slab(A, W, K, 0, m0, n0, asB[0], bsB[0], tid);

  for (int s = 0; s < nslab; ++s) {
    const int cur = s & 1;
    if (s + 1 < nslab) {
      // overlap: DMA next slab into the other buffer while computing this one
      gemm_issue_slab(A, W, K, (s + 1) * 32, m0, n0, asB[cur ^ 1],
                      bsB[cur ^ 1], tid);
      // async loads complete in order: <=4 outstanding => slab s landed
      asm volatile("s_wait_asynccnt 0x4" ::: "memory");
    } else {
      asm volatile("s_wait_asynccnt 0x0" ::: "memory");
    }
    __syncthreads();

    FragAB a[4], b[2];
#pragma unroll
    for (int mt = 0; mt < 4; ++mt) {
      const int m = wr * 64 + mt * 16 + l15;
#pragma unroll
      for (int j = 0; j < 8; ++j)
        a[mt].u[j] = *(const unsigned*)&As[cur][m][a_pair_k(j, half)];
    }
#pragma unroll
    for (int nt = 0; nt < 2; ++nt) {
      const int n = wc * 32 + nt * 16 + l15;
#pragma unroll
      for (int j = 0; j < 8; ++j)
        b[nt].u[j] = *(const unsigned*)&Bs[cur][n][16 * half + 2 * j];
    }
#pragma unroll
    for (int mt = 0; mt < 4; ++mt)
#pragma unroll
      for (int nt = 0; nt < 2; ++nt)
        acc[mt][nt].v = __builtin_amdgcn_wmma_f32_16x16x32_bf16(
            false, a[mt].v, false, b[nt].v, (short)0, acc[mt][nt].v,
            false, false);
    __syncthreads();  // all waves done with buf cur before it is re-DMA'd
  }

  // Epilogue. C layout: VGPR r holds (M = r + 8*half, N = l15).
#pragma unroll
  for (int mt = 0; mt < 4; ++mt)
#pragma unroll
    for (int nt = 0; nt < 2; ++nt)
#pragma unroll
      for (int r = 0; r < 8; ++r) {
        const int m = m0 + wr * 64 + mt * 16 + r + 8 * half;
        const int n = n0 + wc * 32 + nt * 16 + l15;
        const float v = acc[mt][nt].f[r];
        if (OUT_BF16) Obf[(size_t)m * ldo + n] = f2bf(v);
        else          Ofp[(size_t)m * ldo + n] = v;
      }
}

// ---------------------------------------------------------------------------
// In-place RoPE on a bf16 buffer X: [SEQ, nheads*128]. Thread per (s, h, i<64).
__global__ void rope_bf16(unsigned short* __restrict__ X, int nheads) {
  int idx = blockIdx.x * blockDim.x + threadIdx.x;
  int total = SEQ * nheads * 64;
  if (idx >= total) return;
  int i = idx & 63;
  int h = (idx >> 6) % nheads;
  int s = idx / (64 * nheads);
  float freq = __expf(-9.210340371976184f * ((float)i * (1.0f / 64.0f)));
  float ang = (float)s * freq;
  float c, sn;
  __sincosf(ang, &sn, &c);
  unsigned short* p = X + (size_t)s * (nheads * HD) + h * HD;
  float x1 = bf2f(p[i]);
  float x2 = bf2f(p[i + 64]);
  p[i]      = f2bf(x1 * c - x2 * sn);
  p[i + 64] = f2bf(x2 * c + x1 * sn);
}

// ---------------------------------------------------------------------------
// Flash attention, causal, GQA (4 Q heads per KV head).
// grid = (NH, SEQ/64); block = 128 (4 waves). Wave w owns 16 q-rows.
__global__ __launch_bounds__(128)
void flash_attn(const unsigned short* __restrict__ Q,    // [SEQ, 4096]
                const unsigned short* __restrict__ Kb,   // [SEQ, 1024]
                const unsigned short* __restrict__ Vb,   // [SEQ, 1024]
                unsigned short* __restrict__ Ctx) {      // [SEQ, 4096]
  const int h   = blockIdx.x;
  const int qb  = blockIdx.y;
  const int kvh = h >> 2;
  const int s0  = qb * 64;

  const int tid  = threadIdx.x;
  const int wave = tid >> 5;
  const int lane = tid & 31;
  const int half = lane >> 4;
  const int l15  = lane & 15;

  __shared__ unsigned short Qs[64][128];   // [m][d]
  __shared__ unsigned short Ks[64][128];   // [n][d]
  __shared__ unsigned short Vt[128][64];   // [d][n]  (transposed for B frags)
  __shared__ unsigned short Ps[4][16][64]; // per-wave P staging (C->A relayout)
  const unsigned ksBase = lds_off(&Ks[0][0]);

  // Load Q block (64 x 128 bf16).
  for (int i = tid; i < 64 * 16; i += 128) {
    int r = i >> 4, c = i & 15;
    ((uint4*)&Qs[r][0])[c] =
        ((const uint4*)(Q + (size_t)(s0 + r) * HID + h * HD))[c];
  }
  __syncthreads();

  // Preload Q A-fragments (4 k-steps of 32 over d=128).
  FragAB aq[4];
#pragma unroll
  for (int t = 0; t < 4; ++t)
#pragma unroll
    for (int j = 0; j < 8; ++j)
      aq[t].u[j] =
          *(const unsigned*)&Qs[wave * 16 + l15][t * 32 + a_pair_k(j, half)];

  FragC out[8];
  float mrun[8], lrun[8];
#pragma unroll
  for (int dt = 0; dt < 8; ++dt)
#pragma unroll
    for (int r = 0; r < 8; ++r) out[dt].f[r] = 0.f;
#pragma unroll
  for (int r = 0; r < 8; ++r) { mrun[r] = -3.0e38f; lrun[r] = 0.f; }

  const float scale = 0.08838834764831845f;  // 1/sqrt(128)

  for (int kb = 0; kb <= qb; ++kb) {
    const int kbase = kb * 64;
    __syncthreads();  // everyone done with previous Ks/Vt

    // Async-DMA the K block (64x128 bf16 = 1024 16B chunks, 8 rounds).
    {
      const unsigned short* kp = Kb + (size_t)kbase * (NKV * HD);
#pragma unroll
      for (int t = 0; t < 8; ++t) {
        const int chunk = t * 128 + tid;
        const int r = chunk >> 4, c = chunk & 15;
        async_load_b128(ksBase + chunk * 16,
                        (unsigned)((r * (NKV * HD) + kvh * HD + c * 8) * 2),
                        kp);
      }
    }
    // V block, transposed into LDS (VGPR path; transpose needed).
    for (int i = tid; i < 64 * 16; i += 128) {
      int n = i >> 4, c = i & 15;
      uint4 w =
          ((const uint4*)(Vb + (size_t)(kbase + n) * (NKV * HD) + kvh * HD))[c];
      const unsigned short* e = (const unsigned short*)&w;
#pragma unroll
      for (int t = 0; t < 8; ++t) Vt[c * 8 + t][n] = e[t];
    }
    if (kb < qb) {  // gfx1250 global_prefetch of next V slab
      __builtin_prefetch(
          Vb + (size_t)(kbase + 64 + (tid >> 1)) * (NKV * HD) + kvh * HD, 0, 0);
    }
    asm volatile("s_wait_asynccnt 0x0" ::: "memory");
    __syncthreads();

    // S = Q K^T : 4 N-tiles, d accumulated in 4 WMMA steps (paired preloads).
    FragC sc[4];
#pragma unroll
    for (int nt = 0; nt < 4; ++nt) {
#pragma unroll
      for (int r = 0; r < 8; ++r) sc[nt].f[r] = 0.f;
#pragma unroll
      for (int t2 = 0; t2 < 2; ++t2) {
        FragAB bk[2];
#pragma unroll
        for (int p = 0; p < 2; ++p)
#pragma unroll
          for (int j = 0; j < 8; ++j)
            bk[p].u[j] = *(const unsigned*)&Ks[nt * 16 + l15]
                              [(t2 * 2 + p) * 32 + 16 * half + 2 * j];
#pragma unroll
        for (int p = 0; p < 2; ++p)
          sc[nt].v = __builtin_amdgcn_wmma_f32_16x16x32_bf16(
              false, aq[t2 * 2 + p].v, false, bk[p].v, (short)0, sc[nt].v,
              false, false);
      }
    }

    // Online softmax per row (row M = r + 8*half lives in 16 lanes of a half).
#pragma unroll
    for (int r = 0; r < 8; ++r) {
      const int irow = s0 + wave * 16 + r + 8 * half;
      float sv[4];
      float rowmax = -3.0e38f;
#pragma unroll
      for (int nt = 0; nt < 4; ++nt) {
        const int jcol = kbase + nt * 16 + l15;
        float v = sc[nt].f[r] * scale;
        if (jcol > irow) v = -3.0e38f;  // causal mask
        sv[nt] = v;
        rowmax = fmaxf(rowmax, v);
      }
#pragma unroll
      for (int m = 1; m < 16; m <<= 1)
        rowmax = fmaxf(rowmax, __shfl_xor(rowmax, m, 32));
      const float mnew  = fmaxf(mrun[r], rowmax);
      const float alpha = __expf(mrun[r] - mnew);
      float lsum = 0.f;
#pragma unroll
      for (int nt = 0; nt < 4; ++nt) {
        const float p = __expf(sv[nt] - mnew);
        lsum += p;
        Ps[wave][r + 8 * half][nt * 16 + l15] = f2bf(p);
      }
#pragma unroll
      for (int m = 1; m < 16; m <<= 1) lsum += __shfl_xor(lsum, m, 32);
      lrun[r] = lrun[r] * alpha + lsum;
      mrun[r] = mnew;
#pragma unroll
      for (int dt = 0; dt < 8; ++dt) out[dt].f[r] *= alpha;
    }

    // O += P V : A-frags of P via per-wave LDS relayout (same-wave DS ops are
    // program-ordered, no barrier needed); B-frags from Vt.
    FragAB ap[2];
#pragma unroll
    for (int u = 0; u < 2; ++u)
#pragma unroll
      for (int j = 0; j < 8; ++j)
        ap[u].u[j] =
            *(const unsigned*)&Ps[wave][l15][u * 32 + a_pair_k(j, half)];
#pragma unroll
    for (int dt = 0; dt < 8; ++dt) {
      FragAB bv[2];
#pragma unroll
      for (int u = 0; u < 2; ++u)
#pragma unroll
        for (int j = 0; j < 8; ++j)
          bv[u].u[j] =
              *(const unsigned*)&Vt[dt * 16 + l15][u * 32 + 16 * half + 2 * j];
#pragma unroll
      for (int u = 0; u < 2; ++u)
        out[dt].v = __builtin_amdgcn_wmma_f32_16x16x32_bf16(
            false, ap[u].v, false, bv[u].v, (short)0, out[dt].v, false, false);
    }
  }

  // Epilogue: normalize and write context in [S, H] layout.
#pragma unroll
  for (int dt = 0; dt < 8; ++dt)
#pragma unroll
    for (int r = 0; r < 8; ++r) {
      const int row = s0 + wave * 16 + r + 8 * half;
      const int col = h * HD + dt * 16 + l15;
      Ctx[(size_t)row * HID + col] = f2bf(out[dt].f[r] / lrun[r]);
    }
}

// ---------------------------------------------------------------------------
extern "C" void kernel_launch(void* const* d_in, const int* in_sizes, int n_in,
                              void* d_out, int out_size, void* d_ws,
                              size_t ws_size, hipStream_t stream) {
  (void)in_sizes; (void)n_in; (void)out_size; (void)ws_size;
  const float* x  = (const float*)d_in[0];
  const float* wq = (const float*)d_in[1];
  const float* wk = (const float*)d_in[2];
  const float* wv = (const float*)d_in[3];
  const float* wo = (const float*)d_in[4];

  // Workspace carve (bf16 elements). Total ~142.6 MB.
  unsigned short* ws = (unsigned short*)d_ws;
  const size_t SZ_X  = (size_t)SEQ * HID;        // 8 M
  const size_t SZ_WQ = (size_t)HID * HID;        // 16 M
  const size_t SZ_WK = (size_t)NKV * HD * HID;   // 4 M
  unsigned short* xb   = ws;
  unsigned short* wqb  = xb   + SZ_X;
  unsigned short* wkb  = wqb  + SZ_WQ;
  unsigned short* wvb  = wkb  + SZ_WK;
  unsigned short* wob  = wvb  + SZ_WK;
  unsigned short* qbuf = wob  + SZ_WQ;
  unsigned short* kbuf = qbuf + SZ_X;
  unsigned short* vbuf = kbuf + (size_t)SEQ * NKV * HD;
  unsigned short* ctx  = vbuf + (size_t)SEQ * NKV * HD;

  auto cast = [&](const float* s, unsigned short* d, size_t n) {
    int n4 = (int)(n / 4);
    cast_f32_to_bf16<<<(n4 + 255) / 256, 256, 0, stream>>>(s, d, n4);
  };
  cast(x,  xb,  SZ_X);
  cast(wq, wqb, SZ_WQ);
  cast(wk, wkb, SZ_WK);
  cast(wv, wvb, SZ_WK);
  cast(wo, wob, SZ_WQ);

  // Projections: [2048,4096] @ W^T
  dim3 gq(HID / 128, SEQ / 128);        // (32,16)
  dim3 gkv(NKV * HD / 128, SEQ / 128);  // (8,16)
  gemm_a_x_wT<true><<<gq,  256, 0, stream>>>(xb, wqb, qbuf, nullptr, HID, HID);
  gemm_a_x_wT<true><<<gkv, 256, 0, stream>>>(xb, wkb, kbuf, nullptr, HID, NKV * HD);
  gemm_a_x_wT<true><<<gkv, 256, 0, stream>>>(xb, wvb, vbuf, nullptr, HID, NKV * HD);

  // RoPE on Q and K.
  {
    int nq = SEQ * NH * 64;
    rope_bf16<<<(nq + 255) / 256, 256, 0, stream>>>(qbuf, NH);
    int nk = SEQ * NKV * 64;
    rope_bf16<<<(nk + 255) / 256, 256, 0, stream>>>(kbuf, NKV);
  }

  // Flash attention: (head, q-block of 64 rows).
  flash_attn<<<dim3(NH, SEQ / 64), 128, 0, stream>>>(qbuf, kbuf, vbuf, ctx);

  // Output projection -> fp32 d_out.
  gemm_a_x_wT<false><<<gq, 256, 0, stream>>>(ctx, wob, nullptr, (float*)d_out,
                                             HID, HID);
}
